// TransformSinkhornPRModel_56358560858146
// MI455X (gfx1250) — matrix operand order repaired and checked by reading, not compiled, not run on hardware
//
#include <hip/hip_runtime.h>
#include <hip/hip_bf16.h>
#include <math.h>

// ---------------------------------------------------------------------------
// CDNA5 (gfx1250) wave32 WMMA types
// ---------------------------------------------------------------------------
typedef __attribute__((ext_vector_type(16))) _Float16 v16h;
typedef __attribute__((ext_vector_type(8)))  float    v8f;

union FragU { uint4 u[2]; v16h h; };

// A-matrix 16x32 f16 fragment (ISA 7.12.2): lanes 0-15 rows 0-15 hold K{0..7,16..23},
// lanes 16-31 same rows hold K{8..15,24..31}.  Two 16B loads, both 16B aligned.
static __device__ inline v16h load_a_frag(const _Float16* __restrict__ A, int lda,
                                          int row, int k0, int lh) {
  FragU f;
  const _Float16* p = A + (size_t)row * lda + k0 + lh * 8;
  f.u[0] = *(const uint4*)(p);
  f.u[1] = *(const uint4*)(p + 16);
  return f.h;
}

// B-matrix 32x16 f16 fragment: lane = column n, lanes 0-15 hold K{0..7,16..23},
// lanes 16-31 hold K{8..15,24..31}.  B stored row-major over K (NT layout).
static __device__ inline v16h load_b_frag(const _Float16* __restrict__ Y, int ldy,
                                          int colrow, int k0, int lh) {
  FragU f;
  const _Float16* p = Y + (size_t)colrow * ldy + k0 + lh * 8;
  f.u[0] = *(const uint4*)(p);
  f.u[1] = *(const uint4*)(p + 16);
  return f.h;
}

#define WMMA_F16(A, B, C) \
  __builtin_amdgcn_wmma_f32_16x16x32_f16(false, (A), false, (B), (short)0, (C), false, false)

#define JS 4   // j-range split per softmin job (parallelism multiplier)

// ---------------------------------------------------------------------------
// Small helpers
// ---------------------------------------------------------------------------
__global__ void f32_to_f16_k(const float* __restrict__ in, _Float16* __restrict__ out, int n) {
  int i = blockIdx.x * blockDim.x + threadIdx.x;
  if (i < n) out[i] = (_Float16)in[i];
}

// M[768][128] f32 -> Wt[128][768] f16  (so the transform is an NT-GEMM)
__global__ void transpose_f16_k(const float* __restrict__ Min, _Float16* __restrict__ Wt) {
  int i = blockIdx.x * blockDim.x + threadIdx.x;
  if (i < 768 * 128) {
    int k = i / 128, n = i % 128;
    Wt[(size_t)n * 768 + k] = (_Float16)Min[i];
  }
}

__global__ void zero_f32_k(float* __restrict__ p, int n) {
  int i = blockIdx.x * blockDim.x + threadIdx.x;
  if (i < n) p[i] = 0.0f;
}

__global__ void upd_half_k(float* __restrict__ p, const float* __restrict__ pt, int n) {
  int i = blockIdx.x * blockDim.x + threadIdx.x;
  if (i < n) p[i] = 0.5f * (p[i] + pt[i]);
}

// ---------------------------------------------------------------------------
// T = X(rows x 768) @ Wt(128 x 768)^T  via WMMA; writes f16 output + f32 row norms.
// Block = 8 waves; each wave computes 16 rows x 128 cols.
// ---------------------------------------------------------------------------
__global__ void __launch_bounds__(256)
transform_nt(const _Float16* __restrict__ X, const _Float16* __restrict__ Wt,
             _Float16* __restrict__ T, float* __restrict__ norms, int rows) {
  const int lane = threadIdx.x & 31, wv = threadIdx.x >> 5;
  const int lh = lane >> 4, col = lane & 15;
  const int row0 = (blockIdx.x * 8 + wv) * 16;
  if (row0 >= rows) return;
  const int arow = row0 + col;

  float sq[8];
#pragma unroll
  for (int r = 0; r < 8; ++r) sq[r] = 0.0f;

  for (int n = 0; n < 8; ++n) {
    v8f acc = {};
#pragma unroll 4
    for (int k = 0; k < 24; ++k) {
      v16h a = load_a_frag(X, 768, arow, k * 32, lh);
      v16h b = load_b_frag(Wt, 768, n * 16 + col, k * 32, lh);
      acc = WMMA_F16(a, b, acc);
    }
#pragma unroll
    for (int r = 0; r < 8; ++r) {           // C layout: row = lh*8+r, col = n*16+col
      float v = acc[r];
      sq[r] += v * v;
      T[(size_t)(row0 + lh * 8 + r) * 128 + n * 16 + col] = (_Float16)v;
    }
  }
#pragma unroll
  for (int r = 0; r < 8; ++r) {             // reduce |row|^2 across the 16 lanes of a half
    float v = sq[r];
    v += __shfl_xor(v, 1, 16); v += __shfl_xor(v, 2, 16);
    v += __shfl_xor(v, 4, 16); v += __shfl_xor(v, 8, 16);
    if (col == 0) norms[row0 + lh * 8 + r] = v;
  }
}

// ---------------------------------------------------------------------------
// Batched softmin: 8 independent softmin jobs (4 per divergence) in ONE launch,
// each job's j-range split JS ways.  Each wave computes, for 16 rows and one
// j-chunk, the online (m, s) log-sum-exp state of
//   logsumexp_j( logw + hpot[j]/eps - max(.5|x_i|^2 + .5|y_j|^2 - x_i.y_j, 0)/eps )
// with cost tiles recomputed on the fly via WMMA (K = 128).
// ---------------------------------------------------------------------------
struct SoftminJob {
  const _Float16* X; const _Float16* Y;
  const float* nx; const float* ny; const float* hpot;
  float* pm; float* ps;        // partial (m, s): [JS * nxrows] each
  float* outf;                 // per-row result (written by combine kernel)
  float logw;
  int nxrows, nyrows, blk0, row0;   // blk0/row0: prefix offsets for decode
};
struct JobTable { SoftminJob j[8]; int totalBlocks; int totalRows; };

__global__ void __launch_bounds__(256)
softmin_batch(JobTable T, float inv_eps) {
  int jid = 0;
#pragma unroll
  for (int k = 1; k < 8; ++k) if ((int)blockIdx.x >= T.j[k].blk0) jid = k;
  const SoftminJob J = T.j[jid];

  const int local     = (int)blockIdx.x - J.blk0;
  const int rowBlocks = J.nxrows >> 7;           // 128 rows per block
  const int rb = local % rowBlocks;              // row-block index
  const int jc = local / rowBlocks;              // j-chunk index

  const int lane = threadIdx.x & 31, wv = threadIdx.x >> 5;
  const int lh = lane >> 4, col = lane & 15;
  const int row0 = rb * 128 + wv * 16;
  const int arow = row0 + col;
  const int chunk = J.nyrows / JS;
  const int jbeg = jc * chunk, jend = jbeg + chunk;

  const v16h a0 = load_a_frag(J.X, 128, arow, 0,  lh);
  const v16h a1 = load_a_frag(J.X, 128, arow, 32, lh);
  const v16h a2 = load_a_frag(J.X, 128, arow, 64, lh);
  const v16h a3 = load_a_frag(J.X, 128, arow, 96, lh);

  float cx[8], m[8], s[8];
#pragma unroll
  for (int r = 0; r < 8; ++r) {
    cx[r] = 0.5f * J.nx[row0 + lh * 8 + r];
    m[r] = -3.0e38f; s[r] = 0.0f;
  }

  const float logw = J.logw;
  for (int j0 = jbeg; j0 < jend; j0 += 16) {
    const int yrow = j0 + col;
    v16h b0 = load_b_frag(J.Y, 128, yrow, 0,  lh);
    v16h b1 = load_b_frag(J.Y, 128, yrow, 32, lh);
    v16h b2 = load_b_frag(J.Y, 128, yrow, 64, lh);
    v16h b3 = load_b_frag(J.Y, 128, yrow, 96, lh);
    if (j0 + 16 < jend)                       // global_prefetch_b8 of the next tile
      __builtin_prefetch(J.Y + (size_t)(yrow + 16) * 128, 0, 0);

    v8f acc = {};
    acc = WMMA_F16(a0, b0, acc);
    acc = WMMA_F16(a1, b1, acc);
    acc = WMMA_F16(a2, b2, acc);
    acc = WMMA_F16(a3, b3, acc);

    const float hj = logw + J.hpot[yrow] * inv_eps;
    const float cy = 0.5f * J.ny[yrow];
#pragma unroll
    for (int r = 0; r < 8; ++r) {
      float c = fmaxf(cx[r] + cy - acc[r], 0.0f);
      float v = hj - c * inv_eps;
      float d = v - m[r];
      float e = __expf(-fabsf(d));            // single exp per element
      bool  g = d > 0.0f;
      s[r] = g ? (s[r] * e + 1.0f) : (s[r] + e);
      m[r] = g ? v : m[r];
    }
  }

#pragma unroll
  for (int r = 0; r < 8; ++r) {               // combine (m,s) across the 16 lanes of a half
    float mr = m[r], sr = s[r];
#pragma unroll
    for (int mask = 1; mask < 16; mask <<= 1) {
      float mo = __shfl_xor(mr, mask, 16);
      float so = __shfl_xor(sr, mask, 16);
      float mn = fmaxf(mr, mo);
      sr = sr * __expf(mr - mn) + so * __expf(mo - mn);
      mr = mn;
    }
    if (col == 0) {
      const int idx = jc * J.nxrows + row0 + lh * 8 + r;
      J.pm[idx] = mr;
      J.ps[idx] = sr;
    }
  }
}

// Combine the JS partial (m, s) pairs per row -> -eps * (m + log s).
__global__ void __launch_bounds__(256)
combine_k(JobTable T, float eps) {
  const int t = blockIdx.x * blockDim.x + threadIdx.x;
  if (t >= T.totalRows) return;
  int jid = 0;
#pragma unroll
  for (int k = 1; k < 8; ++k) if (t >= T.j[k].row0) jid = k;
  const SoftminJob J = T.j[jid];
  const int r = t - J.row0;

  float m = J.pm[r], s = J.ps[r];
#pragma unroll
  for (int c = 1; c < JS; ++c) {
    float mo = J.pm[c * J.nxrows + r], so = J.ps[c * J.nxrows + r];
    float mn = fmaxf(m, mo);
    s = s * __expf(m - mn) + so * __expf(mo - mn);
    m = mn;
  }
  J.outf[r] = -eps * (m + __logf(s));
}

// ---------------------------------------------------------------------------
// dist = mean(f_xy - f_xx) + mean(g_xy - g_yy); tmp layout [f(4096)|fx(4096)|g(2048)|gy(2048)]
// Single block -> deterministic reduction.
// ---------------------------------------------------------------------------
__global__ void __launch_bounds__(1024) dist_reduce_k(const float* __restrict__ t,
                                                      float* __restrict__ out) {
  __shared__ float sdata[32];
  float acc = 0.0f;
  for (int i = threadIdx.x; i < 6144; i += 1024) {
    if (i < 4096) acc += (t[i] - t[4096 + i]) * (1.0f / 4096.0f);
    else { int j = i - 4096; acc += (t[8192 + j] - t[10240 + j]) * (1.0f / 2048.0f); }
  }
  for (int mask = 16; mask; mask >>= 1) acc += __shfl_xor(acc, mask, 32);
  if ((threadIdx.x & 31) == 0) sdata[threadIdx.x >> 5] = acc;
  __syncthreads();
  if (threadIdx.x < 32) {
    float v = sdata[threadIdx.x];
    for (int mask = 16; mask; mask >>= 1) v += __shfl_xor(v, mask, 32);
    if (threadIdx.x == 0) *out = v;
  }
}

__global__ void finalize_k(const float* __restrict__ scal, float* __restrict__ out, int n) {
  float dd = 0.1f * (scal[1] - scal[0]);      // sigmoid(SF*(dist2 - dist1))
  float r  = 1.0f / (1.0f + __expf(-dd));
  for (int i = threadIdx.x; i < n; i += blockDim.x) out[i] = r;
}

// ---------------------------------------------------------------------------
// Host orchestration
// ---------------------------------------------------------------------------
extern "C" void kernel_launch(void* const* d_in, const int* in_sizes, int n_in,
                              void* d_out, int out_size, void* d_ws, size_t ws_size,
                              hipStream_t stream) {
  (void)in_sizes; (void)n_in; (void)ws_size;
  const float* din  = (const float*)d_in[0];   // d  [4096 x 768]
  const float* s1in = (const float*)d_in[1];   // s1 [2048 x 768]
  const float* s2in = (const float*)d_in[2];   // s2 [2048 x 768]
  const float* Min  = (const float*)d_in[3];   // M  [768 x 128]

  const int N = 4096, Mn = 2048, DI = 768;

  char* w = (char*)d_ws;
  size_t o = 0;
  auto alloc = [&](size_t bytes) { size_t r = o; o += (bytes + 255) & ~(size_t)255; return r; };

  _Float16* d16  = (_Float16*)(w + alloc((size_t)N  * DI * 2));
  _Float16* s116 = (_Float16*)(w + alloc((size_t)Mn * DI * 2));
  _Float16* s216 = (_Float16*)(w + alloc((size_t)Mn * DI * 2));
  _Float16* wt16 = (_Float16*)(w + alloc((size_t)128 * DI * 2));
  _Float16* td16 = (_Float16*)(w + alloc((size_t)N  * 128 * 2));
  _Float16* t116 = (_Float16*)(w + alloc((size_t)Mn * 128 * 2));
  _Float16* t216 = (_Float16*)(w + alloc((size_t)Mn * 128 * 2));
  float*    nd   = (float*)(w + alloc((size_t)N  * 4));
  float*    n1   = (float*)(w + alloc((size_t)Mn * 4));
  float*    n2   = (float*)(w + alloc((size_t)Mn * 4));
  float*    pot1 = (float*)(w + alloc(12288 * 4));   // [f|fx|g|gy]
  float*    pot2 = (float*)(w + alloc(12288 * 4));   // contiguous with pot1
  float*    scal = (float*)(w + alloc(64));          // contiguous with pot2
  float*    tmp1 = (float*)(w + alloc(12288 * 4));
  float*    tmp2 = (float*)(w + alloc(12288 * 4));   // contiguous with tmp1

  // Phase 1: convert inputs to f16, transpose M, transform + norms (WMMA).
  f32_to_f16_k<<<(N * DI + 255) / 256, 256, 0, stream>>>(din, d16, N * DI);
  f32_to_f16_k<<<(Mn * DI + 255) / 256, 256, 0, stream>>>(s1in, s116, Mn * DI);
  f32_to_f16_k<<<(Mn * DI + 255) / 256, 256, 0, stream>>>(s2in, s216, Mn * DI);
  transpose_f16_k<<<(DI * 128 + 255) / 256, 256, 0, stream>>>(Min, wt16);
  transform_nt<<<N / 128, 256, 0, stream>>>(d16,  wt16, td16, nd, N);
  transform_nt<<<Mn / 128, 256, 0, stream>>>(s116, wt16, t116, n1, Mn);
  transform_nt<<<Mn / 128, 256, 0, stream>>>(s216, wt16, t216, n2, Mn);

  // Zero potentials (both divergences) + scalar slots in one contiguous sweep.
  zero_f32_k<<<(24592 + 255) / 256, 256, 0, stream>>>(pot1, 24592);

  // Build the 8-job table (same pointers every iteration; only eps varies).
  const float loga = -logf((float)N);
  const float logb = -logf((float)Mn);
  const int F = 0, FX = 4096, G = 8192, GY = 10240;

  JobTable T;
  {
    const _Float16* Xs[8]   = { td16, t116, td16, t116, td16, t216, td16, t216 };
    const _Float16* Ys[8]   = { t116, td16, td16, t116, t216, td16, td16, t216 };
    const float*    nxs[8]  = { nd, n1, nd, n1, nd, n2, nd, n2 };
    const float*    nys[8]  = { n1, nd, nd, n1, n2, nd, nd, n2 };
    const float*    hps[8]  = { pot1+G, pot1+F, pot1+FX, pot1+GY,
                                pot2+G, pot2+F, pot2+FX, pot2+GY };
    float*          outs[8] = { tmp1+F, tmp1+G, tmp1+FX, tmp1+GY,
                                tmp2+F, tmp2+G, tmp2+FX, tmp2+GY };
    const float     lws[8]  = { logb, loga, loga, logb, logb, loga, loga, logb };
    const int       nxr[8]  = { N, Mn, N, Mn, N, Mn, N, Mn };
    const int       nyr[8]  = { Mn, N, N, Mn, Mn, N, N, Mn };
    int blk = 0, row = 0;
    for (int k = 0; k < 8; ++k) {
      SoftminJob& J = T.j[k];
      J.X = Xs[k]; J.Y = Ys[k]; J.nx = nxs[k]; J.ny = nys[k]; J.hpot = hps[k];
      J.pm = (float*)(w + alloc((size_t)JS * nxr[k] * 4));
      J.ps = (float*)(w + alloc((size_t)JS * nxr[k] * 4));
      J.outf = outs[k]; J.logw = lws[k];
      J.nxrows = nxr[k]; J.nyrows = nyr[k];
      J.blk0 = blk; J.row0 = row;
      blk += (nxr[k] / 128) * JS;
      row += nxr[k];
    }
    T.totalBlocks = blk;   // 768 blocks = 6144 waves per batched launch
    T.totalRows   = row;   // 24576
  }

  // eps schedule (geomloss annealing): 32^2 -> 0.05^2, factor 0.8^2 (30 steps).
  float epsv[64]; int ne = 0;
  { double e = 1024.0; while (e > 0.0025) { epsv[ne++] = (float)e; e *= 0.64; }
    epsv[ne++] = 0.0025f; }

  const int cgrid = (T.totalRows + 255) / 256;
  for (int it = 0; it < ne; ++it) {
    const float eps = epsv[it], ie = 1.0f / eps;
    softmin_batch<<<T.totalBlocks, 256, 0, stream>>>(T, ie);  // all 8 jobs at once
    combine_k<<<cgrid, 256, 0, stream>>>(T, eps);
    // pot1|pot2 and tmp1|tmp2 are contiguous: one fused averaging sweep.
    upd_half_k<<<(24576 + 255) / 256, 256, 0, stream>>>(pot1, tmp1, 24576);
  }

  // Final extrapolation at eps_final with annealed potentials (same job table).
  const float ef = epsv[ne - 1], ief = 1.0f / ef;
  softmin_batch<<<T.totalBlocks, 256, 0, stream>>>(T, ief);
  combine_k<<<cgrid, 256, 0, stream>>>(T, ef);
  dist_reduce_k<<<1, 1024, 0, stream>>>(tmp1, scal + 0);
  dist_reduce_k<<<1, 1024, 0, stream>>>(tmp2, scal + 1);
  finalize_k<<<1, 64, 0, stream>>>(scal, (float*)d_out, out_size);
}
